// GATModel_15401752723913
// MI455X (gfx1250) — matrix operand order repaired and checked by reading, not compiled, not run on hardware
//
#include <hip/hip_runtime.h>
#include <hip/hip_bf16.h>

// ---------------------------------------------------------------------------
// GAT 2-layer forward for MI455X (gfx1250, wave32, WMMA).
// GEMMs via v_wmma_f32_16x16x32_bf16 with bf16 hi/lo split (~fp32 accuracy).
// Each wave computes a 16x64 output strip so the fp32->bf16 A-fragment
// conversion is amortized over 12 WMMAs (4 N-tiles x 3 split products).
// Edge softmax/scatter via fp32 global atomics (h1 is L2-resident: 102MB<192MB).
// ---------------------------------------------------------------------------

#define NNODES 100000
#define NEDGES 1600000
#define ETOT   (NEDGES + NNODES)   // self-loops appended
#define IN_DIM 128
#define HID    64
#define HEADS  4
#define NC     64
#define F1     (HEADS * HID)       // 256
#define NEG_SLOPE 0.2f

typedef __attribute__((ext_vector_type(16))) __bf16 v16bf;
typedef __attribute__((ext_vector_type(8)))  float  v8f;

// ---------------- float <-> order-preserving int (for atomic max) ----------
__device__ __forceinline__ int fenc(float f) {
    int i = __float_as_int(f);
    return (i >= 0) ? i : (i ^ 0x7fffffff);
}
__device__ __forceinline__ float fdec(int k) {
    int i = (k >= 0) ? k : (k ^ 0x7fffffff);
    return __int_as_float(i);
}
__device__ __forceinline__ float lrelu(float v) {
    return v > 0.0f ? v : NEG_SLOPE * v;
}
__device__ __forceinline__ void edge_sd(const int* __restrict__ ei, int e,
                                        int& s, int& d) {
    if (e < NEDGES) { s = ei[e]; d = ei[NEDGES + e]; }
    else            { s = e - NEDGES; d = s; }
}

// ---------------- weight conversion: fp32 -> transposed bf16 hi/lo ---------
// W1 [128,256] -> W1T [256][128]; W2 [256,64] -> W2T [64][256]
__global__ __launch_bounds__(256) void convert_weights_kernel(
    const float* __restrict__ W1, const float* __restrict__ W2,
    __bf16* __restrict__ w1h, __bf16* __restrict__ w1l,
    __bf16* __restrict__ w2h, __bf16* __restrict__ w2l)
{
    int t = blockIdx.x * blockDim.x + threadIdx.x;
    if (t < F1 * IN_DIM) {                 // 256*128
        int n = t >> 7, k = t & 127;
        float f = W1[k * F1 + n];
        __bf16 h = (__bf16)f;
        w1h[t] = h;
        w1l[t] = (__bf16)(f - (float)h);
    } else {
        int u = t - F1 * IN_DIM;
        if (u < NC * F1) {                 // 64*256
            int n = u >> 8, k = u & 255;
            float f = W2[k * NC + n];
            __bf16 h = (__bf16)f;
            w2h[u] = h;
            w2l[u] = (__bf16)(f - (float)h);
        }
    }
}

// ---------------- WMMA GEMM: C[M,NT*16] = A[M,KSTEPS*32] @ Bt^T ------------
// A row-major fp32 (converted to bf16 hi/lo on the fly).
// Bt: [NT*16][KSTEPS*32] bf16 (row = output column), contiguous K.
// One wave per 16 x (16*NTPW) output strip: A fragments amortized over
// NTPW*3 WMMAs per k-step.
template <int KSTEPS, int NTILES, int NTPW>
__global__ __launch_bounds__(256) void gemm_bf16split_kernel(
    const float*  __restrict__ A,
    const __bf16* __restrict__ Bth,
    const __bf16* __restrict__ Btl,
    float* __restrict__ C, int mtiles)
{
    constexpr int K      = KSTEPS * 32;
    constexpr int LDC    = NTILES * 16;
    constexpr int GROUPS = NTILES / NTPW;

    const int lane = threadIdx.x & 31;
    const int wave = blockIdx.x * (blockDim.x >> 5) + (threadIdx.x >> 5);
    if (wave >= mtiles * GROUPS) return;
    const int mt = wave / GROUPS;
    const int gp = wave - mt * GROUPS;
    const int ntbase = gp * NTPW;

    const int l15 = lane & 15;
    const int kb0 = (lane < 16) ? 0 : 16;   // A: K base (16 contiguous)
    const int kbB = (lane < 16) ? 0 : 16;   // B: K base (16 contiguous)

    const float* arow = A + (size_t)(mt * 16 + l15) * K;
    const size_t bofs = (size_t)l15 * K + kbB;

    v8f acc[NTPW];
#pragma unroll
    for (int j = 0; j < NTPW; ++j)
        acc[j] = (v8f){0.f, 0.f, 0.f, 0.f, 0.f, 0.f, 0.f, 0.f};

#pragma unroll
    for (int ks = 0; ks < KSTEPS; ++ks) {
        // ---- A fragment (hi/lo bf16 split), reused for all NTPW tiles ----
        const float* p = arow + ks * 32;
        v16bf ah, al;
#pragma unroll
        for (int i = 0; i < 8; ++i) {
            float f0 = p[kb0 + i];
            float f1 = p[kb0 + 8 + i];
            __bf16 h0 = (__bf16)f0;
            __bf16 h1 = (__bf16)f1;
            ah[i]     = h0;
            ah[8 + i] = h1;
            al[i]     = (__bf16)(f0 - (float)h0);
            al[8 + i] = (__bf16)(f1 - (float)h1);
        }
#pragma unroll
        for (int j = 0; j < NTPW; ++j) {
            const __bf16* bh = Bth + (size_t)(ntbase + j) * 16 * K + bofs + ks * 32;
            const __bf16* bl = Btl + (size_t)(ntbase + j) * 16 * K + bofs + ks * 32;
            v16bf bhv = *(const v16bf*)bh;
            v16bf blv = *(const v16bf*)bl;
            acc[j] = __builtin_amdgcn_wmma_f32_16x16x32_bf16(
                false, ah, false, bhv, (short)0, acc[j], false, false);
            acc[j] = __builtin_amdgcn_wmma_f32_16x16x32_bf16(
                false, ah, false, blv, (short)0, acc[j], false, false);
            acc[j] = __builtin_amdgcn_wmma_f32_16x16x32_bf16(
                false, al, false, bhv, (short)0, acc[j], false, false);
        }
    }

    const int rbase = (lane >> 4) * 8;     // C: lanes 16-31 hold rows 8..15
#pragma unroll
    for (int j = 0; j < NTPW; ++j) {
        float* crow = C + (size_t)(mt * 16 + rbase) * LDC + (ntbase + j) * 16 + l15;
#pragma unroll
        for (int r = 0; r < 8; ++r) crow[(size_t)r * LDC] = acc[j][r];
    }
}

// ---------------- per-node attention logits + segment-max init ------------
template <int H>
__global__ __launch_bounds__(256) void logits_kernel(
    const float* __restrict__ h, const float* __restrict__ a_s,
    const float* __restrict__ a_d, float* __restrict__ als,
    float* __restrict__ ald, int* __restrict__ menc)
{
    int t = blockIdx.x * blockDim.x + threadIdx.x;
    if (t >= NNODES * H) return;
    int node = t / H, hd = t - node * H;
    const float* row = h + (size_t)node * (H * 64) + hd * 64;
    const float* as  = a_s + hd * 64;
    const float* ad  = a_d + hd * 64;
    float s0 = 0.f, s1 = 0.f;
#pragma unroll 8
    for (int c = 0; c < 64; ++c) {
        float v = row[c];
        s0 += v * as[c];
        s1 += v * ad[c];
    }
    als[t]  = s0;
    ald[t]  = s1;
    menc[t] = fenc(-3.0e38f);
}

// ---------------- edge pass A: segment max (atomic int-max) ---------------
template <int H>
__global__ __launch_bounds__(256) void edge_max_kernel(
    const int* __restrict__ ei, const float* __restrict__ als,
    const float* __restrict__ ald, int* __restrict__ menc)
{
    int t = blockIdx.x * blockDim.x + threadIdx.x;
    if (t >= ETOT * H) return;
    int e = t / H, hd = t - e * H;
    int s, d;
    edge_sd(ei, e, s, d);
    float v = lrelu(als[s * H + hd] + ald[d * H + hd]);
    atomicMax(menc + d * H + hd, fenc(v));
}

// ---------------- edge pass B: exp(e-m), segment sum ----------------------
template <int H>
__global__ __launch_bounds__(256) void edge_expsum_kernel(
    const int* __restrict__ ei, const float* __restrict__ als,
    const float* __restrict__ ald, const int* __restrict__ menc,
    float* __restrict__ exbuf, float* __restrict__ den)
{
    int t = blockIdx.x * blockDim.x + threadIdx.x;
    if (t >= ETOT * H) return;
    int e = t / H, hd = t - e * H;
    int s, d;
    edge_sd(ei, e, s, d);
    float v  = lrelu(als[s * H + hd] + ald[d * H + hd]);
    float ex = __expf(v - fdec(menc[d * H + hd]));
    exbuf[t] = ex;
    atomicAdd(den + d * H + hd, ex);
}

// ---------------- edge pass C: weighted scatter-add of messages -----------
// One thread per (edge, float4-of-features). Q = H*16 float4s per edge.
template <int H>
__global__ __launch_bounds__(256) void edge_scatter_kernel(
    const int* __restrict__ ei, const float* __restrict__ exbuf,
    const float* __restrict__ den, const float* __restrict__ hfeat,
    float* __restrict__ out)
{
    constexpr int Q = H * 16;
    int t = blockIdx.x * blockDim.x + threadIdx.x;
    if (t >= ETOT * Q) return;
    int e = t / Q, q = t - e * Q;
    int hd = q >> 4;
    int s, d;
    edge_sd(ei, e, s, d);
    float alpha = exbuf[e * H + hd] / den[d * H + hd];
    const float4* hp = (const float4*)hfeat;
    float4 v = hp[(size_t)s * Q + q];
    float* o = out + ((size_t)d * Q + q) * 4;
    atomicAdd(o + 0, alpha * v.x);
    atomicAdd(o + 1, alpha * v.y);
    atomicAdd(o + 2, alpha * v.z);
    atomicAdd(o + 3, alpha * v.w);
}

// ---------------- bias (+ optional relu) ----------------------------------
__global__ __launch_bounds__(256) void bias_act_kernel(
    float* __restrict__ out, const float* __restrict__ b,
    int total, int colmask, int do_relu)
{
    int t = blockIdx.x * blockDim.x + threadIdx.x;
    if (t >= total) return;
    float v = out[t] + b[t & colmask];
    out[t]  = do_relu ? fmaxf(v, 0.0f) : v;
}

// ---------------------------------------------------------------------------
extern "C" void kernel_launch(void* const* d_in, const int* in_sizes, int n_in,
                              void* d_out, int out_size, void* d_ws, size_t ws_size,
                              hipStream_t stream)
{
    const float* x      = (const float*)d_in[0];
    const int*   ei     = (const int*)  d_in[1];
    const float* W1     = (const float*)d_in[2];
    const float* a_src1 = (const float*)d_in[3];
    const float* a_dst1 = (const float*)d_in[4];
    const float* b1     = (const float*)d_in[5];
    const float* W2     = (const float*)d_in[6];
    const float* a_src2 = (const float*)d_in[7];
    const float* a_dst2 = (const float*)d_in[8];
    const float* b2     = (const float*)d_in[9];
    float* out = (float*)d_out;

    // ---- workspace carve ----
    char* base = (char*)d_ws;
    size_t off = 0;
    auto alloc = [&](size_t bytes) -> char* {
        char* p = base + off;
        off += (bytes + 255) & ~(size_t)255;
        return p;
    };
    float*  h1   = (float*)alloc((size_t)NNODES * F1 * 4);   // x@W1 ; later reused as h2
    float*  out1 = (float*)alloc((size_t)NNODES * F1 * 4);   // layer-1 aggregation
    float*  ex1  = (float*)alloc((size_t)ETOT * HEADS * 4);  // later reused as ex2
    float*  al1s = (float*)alloc((size_t)NNODES * HEADS * 4);
    float*  al1d = (float*)alloc((size_t)NNODES * HEADS * 4);
    float*  den1 = (float*)alloc((size_t)NNODES * HEADS * 4);
    int*    m1   = (int*)  alloc((size_t)NNODES * HEADS * 4);
    float*  al2s = (float*)alloc((size_t)NNODES * 4);
    float*  al2d = (float*)alloc((size_t)NNODES * 4);
    float*  den2 = (float*)alloc((size_t)NNODES * 4);
    int*    m2   = (int*)  alloc((size_t)NNODES * 4);
    __bf16* w1h  = (__bf16*)alloc((size_t)F1 * IN_DIM * 2);
    __bf16* w1l  = (__bf16*)alloc((size_t)F1 * IN_DIM * 2);
    __bf16* w2h  = (__bf16*)alloc((size_t)NC * F1 * 2);
    __bf16* w2l  = (__bf16*)alloc((size_t)NC * F1 * 2);
    float*  h2   = h1;    // alias: h1 dead after layer-1 scatter
    float*  ex2  = ex1;   // alias: ex1 dead after layer-1 scatter

    auto nb = [](long long n) { return (int)((n + 255) / 256); };

    // ---- zero accumulators (graph-capturable stream memsets) ----
    hipMemsetAsync(out1, 0, (size_t)NNODES * F1 * 4, stream);
    hipMemsetAsync(den1, 0, (size_t)NNODES * HEADS * 4, stream);
    hipMemsetAsync(den2, 0, (size_t)NNODES * 4, stream);
    hipMemsetAsync(out,  0, (size_t)NNODES * NC * 4, stream);

    // ---- weights -> transposed bf16 hi/lo ----
    convert_weights_kernel<<<nb(F1 * IN_DIM + NC * F1), 256, 0, stream>>>(
        W1, W2, w1h, w1l, w2h, w2l);

    // ---- layer 1 ----
    {   // GEMM1: h1 = x @ W1   (6250 m-tiles x 4 n-groups, 8 waves/block)
        int mtiles = NNODES / 16;
        int waves  = mtiles * (F1 / 16 / 4);
        gemm_bf16split_kernel<IN_DIM / 32, F1 / 16, 4>
            <<<(waves + 7) / 8, 256, 0, stream>>>(x, w1h, w1l, h1, mtiles);
    }
    logits_kernel<HEADS><<<nb((long long)NNODES * HEADS), 256, 0, stream>>>(
        h1, a_src1, a_dst1, al1s, al1d, m1);
    edge_max_kernel<HEADS><<<nb((long long)ETOT * HEADS), 256, 0, stream>>>(
        ei, al1s, al1d, m1);
    edge_expsum_kernel<HEADS><<<nb((long long)ETOT * HEADS), 256, 0, stream>>>(
        ei, al1s, al1d, m1, ex1, den1);
    edge_scatter_kernel<HEADS><<<nb((long long)ETOT * HEADS * 16), 256, 0, stream>>>(
        ei, ex1, den1, h1, out1);
    bias_act_kernel<<<nb((long long)NNODES * F1), 256, 0, stream>>>(
        out1, b1, NNODES * F1, F1 - 1, 1);

    // ---- layer 2 ----
    {   // GEMM2: h2 = relu(out1) @ W2   (6250 m-tiles, one 16x64 strip each)
        int mtiles = NNODES / 16;
        int waves  = mtiles;   // NTILES=4, NTPW=4 -> 1 group
        gemm_bf16split_kernel<F1 / 32, NC / 16, 4>
            <<<(waves + 7) / 8, 256, 0, stream>>>(out1, w2h, w2l, h2, mtiles);
    }
    logits_kernel<1><<<nb((long long)NNODES), 256, 0, stream>>>(
        h2, a_src2, a_dst2, al2s, al2d, m2);
    edge_max_kernel<1><<<nb((long long)ETOT), 256, 0, stream>>>(
        ei, al2s, al2d, m2);
    edge_expsum_kernel<1><<<nb((long long)ETOT), 256, 0, stream>>>(
        ei, al2s, al2d, m2, ex2, den2);
    edge_scatter_kernel<1><<<nb((long long)ETOT * 16), 256, 0, stream>>>(
        ei, ex2, den2, h2, out);
    bias_act_kernel<<<nb((long long)NNODES * NC), 256, 0, stream>>>(
        out, b2, NNODES * NC, NC - 1, 0);

    (void)in_sizes; (void)n_in; (void)out_size; (void)ws_size;
}
